// PopPredict_2027224564328
// MI455X (gfx1250) — matrix operand order repaired and checked by reading, not compiled, not run on hardware
//
#include <hip/hip_runtime.h>
#include <hip/hip_bf16.h>
#include <math.h>

typedef __attribute__((ext_vector_type(16))) _Float16 v16h;
typedef __attribute__((ext_vector_type(8)))  float    v8f;

#define DDIM 128
#define TDIM 4096

// ---------------------------------------------------------------------------
// Kernel A: EMA prefix scan per batch row (one wave32 per row), gather e[tb],
// sigmoid -> d_out[0..B).  e_0 = p_0 ; e_t = 0.9 e_{t-1} + 0.1 p_t.
//
// Blocked affine scan: each lane serially composes an 8-element segment
// (A,b) with comb((A1,b1),(A2,b2)) = (A1*A2, A2*b1 + b2), then a 5-step
// Hillis-Steele lane scan covers 256 elements per wave iteration. Carry is
// one affine apply per iteration. Final block: owning lane re-walks its
// segment to extract e[tb] exactly.
// ---------------------------------------------------------------------------
__global__ void __launch_bounds__(256)
ema_pop_kernel(const float* __restrict__ ph, const int* __restrict__ times,
               float* __restrict__ out_pop, int Bn) {
  const int wave = (int)((blockIdx.x * blockDim.x + threadIdx.x) >> 5);
  const int lane = (int)(threadIdx.x & 31);
  if (wave >= Bn) return;

  int tb = times[wave] - 1;
  if (tb < 0) tb = 0;
  if (tb > TDIM - 1) tb = TDIM - 1;

  const float* rowp = ph + (size_t)wave * TDIM;
  const float cc = 0.9f, aa = 0.1f;

  float carry = 0.0f;               // e at last element of previous block
  float result = 0.0f;
  const int nblocks = (tb >> 8) + 1;   // 256 elements per block

  for (int blk = 0; blk < nblocks; ++blk) {
    const int t0 = (blk << 8) + (lane << 3);     // this lane's 8-elem segment
    // prefetch one full iteration (1 KB/wave) ahead; speculative, OOB-safe
    __builtin_prefetch(rowp + t0 + 256, 0, 1);

    const float4 q0 = *(const float4*)(rowp + t0);
    const float4 q1 = *(const float4*)(rowp + t0 + 4);
    float pv[8] = {q0.x, q0.y, q0.z, q0.w, q1.x, q1.y, q1.z, q1.w};

    // serial affine composition of this lane's segment
    float A = 1.0f, b = 0.0f;
    #pragma unroll
    for (int k = 0; k < 8; ++k) {
      const bool first = (t0 + k) == 0;
      const float ae = first ? 0.0f : cc;
      const float be = first ? pv[k] : aa * pv[k];
      b = ae * b + be;
      A = ae * A;
    }

    // inclusive Hillis-Steele scan of segment affines over 32 lanes
    float Ai = A, bi = b;
    #pragma unroll
    for (int d = 1; d < 32; d <<= 1) {
      const float Ap = __shfl_up(Ai, d);
      const float bp = __shfl_up(bi, d);
      if (lane >= d) { bi = Ai * bp + bi; Ai = Ai * Ap; }
    }

    if (blk == nblocks - 1) {
      // prefix value just before this lane's segment
      const float Apv = __shfl_up(Ai, 1);
      const float bpv = __shfl_up(bi, 1);
      const float eprev = (lane == 0) ? carry : (Apv * carry + bpv);
      // re-walk segment; owning lane ends with res = e[tb]
      float e = eprev, res = eprev;
      #pragma unroll
      for (int k = 0; k < 8; ++k) {
        const bool first = (t0 + k) == 0;
        const float ae = first ? 0.0f : cc;
        const float be = first ? pv[k] : aa * pv[k];
        e = ae * e + be;
        if (t0 + k <= tb) res = e;
      }
      result = __shfl(res, (tb >> 3) & 31);
    }
    carry = __shfl(Ai, 31) * carry + __shfl(bi, 31);
  }
  if (lane == 0) out_pop[wave] = 1.0f / (1.0f + expf(-result));
}

// ---------------------------------------------------------------------------
// Prep kernel: bake folded weight vectors into f16 WMMA B-fragments in d_ws.
// pass 0: item  -> U1 = Wt[128..255]              (B col 0)
// pass 1: time  -> U2 = Wt[256..383] - Wt[0..127] (B col 0)  (gap fold)
// pass 2: rel   -> U3 = Wt[384..511] + Wt[0..127] (B col 0)
// pass 3: cat   -> Ws1 = Ws[128..255]             (B col 1)
// pass 4: store -> Ws2 = Ws[256..383]             (B col 1)
// B 32x16 f16 layout: lane l holds column N=l%16; half h -> K = (l<16?0:8) + (h<8 ? h : h+8).
// Also: scalars c1 = <W_rating, Ws[0:128]>, c0 = <b_rating, Ws[0:128]>.
// ---------------------------------------------------------------------------
__global__ void prep_weights_kernel(const float* __restrict__ W_time,
                                    const float* __restrict__ W_side,
                                    const float* __restrict__ W_rating,
                                    const float* __restrict__ b_rating,
                                    _Float16* __restrict__ wsB,
                                    float* __restrict__ wsScal) {
  const int tid = (int)(blockIdx.x * blockDim.x + threadIdx.x);
  if (tid < 5 * 4 * 32) {
    const int frag = tid >> 5;          // 0..19  = p*4 + c
    const int lane = tid & 31;
    const int p = frag >> 2;
    const int c = frag & 3;
    const int N = lane & 15;
    const int kbase = c * 32 + ((lane & 16) ? 8 : 0);
    const int target = (p < 3) ? 0 : 1;
    _Float16* dst = wsB + (size_t)tid * 16;
    #pragma unroll
    for (int h = 0; h < 16; ++h) {
      float v = 0.0f;
      if (N == target) {
        const int k = kbase + ((h < 8) ? h : h + 8);
        if      (p == 0) v = W_time[128 + k];
        else if (p == 1) v = W_time[256 + k] - W_time[k];
        else if (p == 2) v = W_time[384 + k] + W_time[k];
        else if (p == 3) v = W_side[128 + k];
        else             v = W_side[256 + k];
      }
      dst[h] = (_Float16)v;
    }
  }
  if (tid == 0) {
    float c1 = 0.0f, c0 = 0.0f;
    for (int k = 0; k < DDIM; ++k) {
      c1 += W_rating[k] * W_side[k];
      c0 += b_rating[k] * W_side[k];
    }
    wsScal[0] = c1;
    wsScal[1] = c0;
  }
}

// ---------------------------------------------------------------------------
// Kernel B: fused side/time/attention. One wave per 16 batch rows.
// 5 gathered-embedding passes x 4 K-steps of v_wmma_f32_16x16x32_f16,
// accumulating time parts in C column 0 and side parts in column 1.
// ---------------------------------------------------------------------------
__device__ inline v16h load_frag_a(const float* __restrict__ rowp, int k0, int base1) {
  // A 16x32 f16 layout: lane<16 -> h:0..7 => K=h,  h:8..15 => K=h+8
  //                     lane>=16 adds +8 to both ranges (base1 = 8)
  const float* p = rowp + k0 + base1;
  const float4 x0 = *(const float4*)(p);
  const float4 x1 = *(const float4*)(p + 4);
  const float4 x2 = *(const float4*)(p + 16);
  const float4 x3 = *(const float4*)(p + 20);
  v16h a;
  a[0]  = (_Float16)x0.x; a[1]  = (_Float16)x0.y; a[2]  = (_Float16)x0.z; a[3]  = (_Float16)x0.w;
  a[4]  = (_Float16)x1.x; a[5]  = (_Float16)x1.y; a[6]  = (_Float16)x1.z; a[7]  = (_Float16)x1.w;
  a[8]  = (_Float16)x2.x; a[9]  = (_Float16)x2.y; a[10] = (_Float16)x2.z; a[11] = (_Float16)x2.w;
  a[12] = (_Float16)x3.x; a[13] = (_Float16)x3.y; a[14] = (_Float16)x3.z; a[15] = (_Float16)x3.w;
  return a;
}

__global__ void __launch_bounds__(128)
fuse_kernel(const float* __restrict__ rating,
            const int* __restrict__ item_ids, const int* __restrict__ times,
            const int* __restrict__ rel_times, const int* __restrict__ cats,
            const int* __restrict__ stores,
            const float* __restrict__ item_emb, const float* __restrict__ cat_emb,
            const float* __restrict__ store_emb, const float* __restrict__ time_emb,
            const float* __restrict__ b_side, const float* __restrict__ b_time,
            const float* __restrict__ att_w,
            const _Float16* __restrict__ wsB, const float* __restrict__ wsScal,
            float* __restrict__ d_out, int Bn) {
  __shared__ float sC[4][2][16];

  const int waveInBlk = (int)(threadIdx.x >> 5);
  const int lane      = (int)(threadIdx.x & 31);
  const int tile      = (int)blockIdx.x * 4 + waveInBlk;     // grid sized exactly
  const int r         = lane & 15;
  const int row       = (tile << 4) + r;
  const int base1     = (lane & 16) ? 8 : 0;

  const float* srcs[5];
  srcs[0] = item_emb  + (size_t)item_ids[row]  * DDIM;
  srcs[1] = time_emb  + (size_t)times[row]     * DDIM;
  srcs[2] = time_emb  + (size_t)rel_times[row] * DDIM;
  srcs[3] = cat_emb   + (size_t)cats[row]      * DDIM;
  srcs[4] = store_emb + (size_t)stores[row]    * DDIM;

  v8f C = {};
  #pragma unroll
  for (int p = 0; p < 5; ++p) {
    #pragma unroll
    for (int c = 0; c < 4; ++c) {
      const v16h a = load_frag_a(srcs[p], c * 32, base1);
      const v16h b = *(const v16h*)(wsB + (size_t)((p * 4 + c) * 32 + lane) * 16);
      C = __builtin_amdgcn_wmma_f32_16x16x32_f16(false, a, false, b,
                                                 (short)0, C, false, false);
    }
  }

  // C layout: lane 0-15 -> N=lane, M=vgpr; lane 16-31 -> N=lane-16, M=vgpr+8
  const int N = lane & 15;
  const int mbase = (lane & 16) ? 8 : 0;
  if (N < 2) {
    #pragma unroll
    for (int v = 0; v < 8; ++v) sC[waveInBlk][N][mbase + v] = C[v];
  }
  __syncthreads();

  if (lane < 16) {
    // ModuleTime: LeakyReLU(0.01) then sigmoid
    const float tpre = sC[waveInBlk][0][r] + b_time[0];
    const float tl = (tpre >= 0.0f) ? tpre : 0.01f * tpre;
    const float time_out = 1.0f / (1.0f + expf(-tl));
    // ModuleSideInfo: cat/store dots + folded rating terms
    const float spre = sC[waveInBlk][1][r] + rating[row] * wsScal[0] + wsScal[1] + b_side[0];
    const float side_out = 1.0f / (1.0f + expf(-spre));
    // pop output produced by kernel A earlier on this stream
    const float pop = d_out[row];
    // softmax over att_w (3 values)
    const float a0 = att_w[0], a1 = att_w[1], a2 = att_w[2];
    const float mx = fmaxf(a0, fmaxf(a1, a2));
    const float e0 = expf(a0 - mx), e1 = expf(a1 - mx), e2 = expf(a2 - mx);
    const float inv = 1.0f / (e0 + e1 + e2);
    const float outv = (e0 * pop + e1 * time_out + e2 * side_out) * inv;

    d_out[Bn + row]     = time_out;
    d_out[2 * Bn + row] = side_out;
    d_out[3 * Bn + row] = outv;
  }
}

// ---------------------------------------------------------------------------
extern "C" void kernel_launch(void* const* d_in, const int* in_sizes, int n_in,
                              void* d_out, int out_size, void* d_ws, size_t ws_size,
                              hipStream_t stream) {
  const float* pop_history = (const float*)d_in[0];
  const float* rating      = (const float*)d_in[1];
  const int*   item_ids    = (const int*)d_in[2];
  const int*   times       = (const int*)d_in[3];
  const int*   rel_times   = (const int*)d_in[4];
  const int*   cats        = (const int*)d_in[5];
  const int*   stores      = (const int*)d_in[6];
  const float* item_emb    = (const float*)d_in[7];
  const float* cat_emb     = (const float*)d_in[8];
  const float* store_emb   = (const float*)d_in[9];
  const float* time_emb    = (const float*)d_in[10];
  const float* W_rating    = (const float*)d_in[11];
  const float* b_rating    = (const float*)d_in[12];
  const float* W_side      = (const float*)d_in[13];
  const float* b_side      = (const float*)d_in[14];
  const float* W_time      = (const float*)d_in[15];
  const float* b_time      = (const float*)d_in[16];
  const float* att_w       = (const float*)d_in[17];

  float* out = (float*)d_out;
  const int Bn = in_sizes[1];               // rating_number has B elements

  _Float16* wsB    = (_Float16*)d_ws;                   // 5*4*32*16 halfs = 20480 B
  float*    wsScal = (float*)((char*)d_ws + 20480);     // 2 floats

  // Prep weights (independent of EMA)
  prep_weights_kernel<<<5, 128, 0, stream>>>(W_time, W_side, W_rating, b_rating,
                                             wsB, wsScal);

  // EMA scan: one wave32 per row -> pop_hist_out in d_out[0..B)
  const int wavesPerBlk = 8;                 // 256 threads
  const int nBlkA = (Bn + wavesPerBlk - 1) / wavesPerBlk;
  ema_pop_kernel<<<nBlkA, 256, 0, stream>>>(pop_history, times, out, Bn);

  // Fused WMMA side/time/attention: one wave per 16 rows, 4 waves per block
  const int tiles = Bn / 16;                 // 512
  fuse_kernel<<<tiles / 4, 128, 0, stream>>>(rating, item_ids, times, rel_times,
                                             cats, stores, item_emb, cat_emb,
                                             store_emb, time_emb, b_side, b_time,
                                             att_w, wsB, wsScal, out, Bn);
}
